// HANLinkPredModel_31628139167801
// MI455X (gfx1250) — matrix operand order repaired and checked by reading, not compiled, not run on hardware
//
#include <hip/hip_runtime.h>
#include <hip/hip_bf16.h>
#include <stdint.h>

// ---------------------------------------------------------------------------
// HAN link-pred model for MI455X (gfx1250, wave32, WMMA).
// Dense projections: bf16 WMMA (v_wmma_f32_16x16x32_bf16), f32 accumulate.
// bf16-input GEMMs stage X tiles via async global->LDS (ASYNCcnt) when the
// tile is full; weights pre-converted to bf16 once per launch.
// Edge softmax: 3-pass atomic segment ops (ordered-int float max, f32 adds).
// ---------------------------------------------------------------------------

typedef __attribute__((ext_vector_type(16))) __bf16 v16bf;
typedef __attribute__((ext_vector_type(8)))  float  v8f;

#define NA 50000
#define NP 100000
#define EW 1600000
#define EC 3200000

__device__ __forceinline__ unsigned short f2bf(float f) {
  unsigned int x = __builtin_bit_cast(unsigned int, f);
  x += 0x7FFFu + ((x >> 16) & 1u);               // round-to-nearest-even
  return (unsigned short)(x >> 16);
}
__device__ __forceinline__ unsigned int pk2bf(float a, float b) {
  return (unsigned int)f2bf(a) | ((unsigned int)f2bf(b) << 16);
}
// order-preserving float<->int mapping for integer atomicMax on floats
__device__ __forceinline__ int f2oi(float f) {
  int i = __builtin_bit_cast(int, f);
  return i >= 0 ? i : (i ^ 0x7FFFFFFF);
}
__device__ __forceinline__ float oi2f(int i) {
  return __builtin_bit_cast(float, i >= 0 ? i : (i ^ 0x7FFFFFFF));
}

union Frag { unsigned int u[8]; v16bf v; };

// ---------------------------------------------------------------------------
// Y[N,64] = act(X[N,K] @ W[64,K]^T + b)   (bf16 WMMA, f32 accumulate)
// Block: 128 threads = 4 waves; each wave computes a 16x64 tile (4 n-tiles).
// LDS pitch K+8 shorts: 16B-aligned rows (async b128) and conflict-free
// fragment reads (row stride 36/68/132 dwords, odd multiple of 4).
// ---------------------------------------------------------------------------
template<int K, bool IN_BF16, bool BIAS_RELU, bool OUT_BF16>
__global__ __launch_bounds__(128) void gemm_wmma(
    const void* __restrict__ Xv, const unsigned short* __restrict__ Wb,
    const float* __restrict__ bias, void* __restrict__ Yv, int nrows)
{
  constexpr int MT = 64;
  constexpr int LP = K + 8;
  __shared__ unsigned short sX[MT * LP];

  const int tid  = threadIdx.x;
  const int m0   = blockIdx.x * MT;
  const bool full = (m0 + MT <= nrows);

  if (IN_BF16) {
    if (full) {
      // async global->LDS staging: 16B chunks, no VGPR round-trip (ASYNCcnt)
      constexpr int CPR = K / 8;                 // 16B chunks per row
      const char* gbase = (const char*)Xv + (size_t)m0 * K * 2;
      for (int c = tid; c < MT * CPR; c += 128) {
        int row = c / CPR, ch = c % CPR;
        unsigned int lds = (unsigned int)(uintptr_t)&sX[row * LP] + (unsigned)(ch * 16);
        unsigned int go  = (unsigned int)(row * (K * 2) + ch * 16);
        asm volatile("global_load_async_to_lds_b128 %0, %1, %2"
                     :: "v"(lds), "v"(go), "s"(gbase) : "memory");
      }
      asm volatile("s_wait_asynccnt 0" ::: "memory");
    } else {
      const unsigned int* X = (const unsigned int*)Xv;
      for (int i = tid; i < MT * (K / 2); i += 128) {
        int r = i / (K / 2), c2 = i % (K / 2);
        int gr = m0 + r;
        unsigned int val = (gr < nrows) ? X[(size_t)gr * (K / 2) + c2] : 0u;
        *(unsigned int*)&sX[r * LP + 2 * c2] = val;
      }
    }
  } else {
    const float2* X = (const float2*)Xv;
    for (int i = tid; i < MT * (K / 2); i += 128) {
      int r = i / (K / 2), c2 = i % (K / 2);
      int gr = m0 + r;
      unsigned int val = 0u;
      if (gr < nrows) { float2 xv = X[(size_t)gr * (K / 2) + c2]; val = pk2bf(xv.x, xv.y); }
      *(unsigned int*)&sX[r * LP + 2 * c2] = val;
    }
  }
  __syncthreads();

  const int wave = tid >> 5, lane = tid & 31;
  const int hi   = lane >> 4;                    // lane-half (0/1)
  const int lrow = (wave << 4) + (lane & 15);    // A-frag row within tile
  v8f cfr[4] = {};

  for (int kb = 0; kb < K; kb += 32) {
    Frag a;
#pragma unroll
    for (int v = 0; v < 8; ++v) {
      // ISA 16-bit A 16x32 layout: k = (v>=4)*16 + (lane>=16)*8 + (v&3)*2
      int ko = ((v & 4) << 2) + (hi << 3) + ((v & 3) << 1);
      a.u[v] = *(const unsigned int*)&sX[lrow * LP + kb + ko];
    }
#pragma unroll
    for (int nt = 0; nt < 4; ++nt) {
      Frag b;
      const int n = (nt << 4) + (lane & 15);
#pragma unroll
      for (int v = 0; v < 8; ++v) {
        // B 32x16 layout: k = (v>=4)*16 + (lane>=16)*8 + (v&3)*2, col n = lane&15
        int ko = ((v & 4) << 2) + (hi << 3) + ((v & 3) << 1);
        b.u[v] = *(const unsigned int*)&Wb[(size_t)n * K + kb + ko];
      }
      cfr[nt] = __builtin_amdgcn_wmma_f32_16x16x32_bf16(
          false, a.v, false, b.v, (short)0, cfr[nt], false, false);
    }
  }

  if (full) {                                    // unguarded epilogue (common)
#pragma unroll
    for (int nt = 0; nt < 4; ++nt) {
      const int n = (nt << 4) + (lane & 15);
      float bv = BIAS_RELU ? bias[n] : 0.f;
#pragma unroll
      for (int r = 0; r < 8; ++r) {
        int gr = m0 + (wave << 4) + r + (hi << 3);   // C/D: M = r + 8*half
        float val = cfr[nt][r];
        if (BIAS_RELU) { val += bv; val = val > 0.f ? val : 0.f; }
        if (OUT_BF16) ((unsigned short*)Yv)[(size_t)gr * 64 + n] = f2bf(val);
        else          ((float*)Yv)[(size_t)gr * 64 + n] = val;
      }
    }
  } else {
#pragma unroll
    for (int nt = 0; nt < 4; ++nt) {
      const int n = (nt << 4) + (lane & 15);
      float bv = BIAS_RELU ? bias[n] : 0.f;
#pragma unroll
      for (int r = 0; r < 8; ++r) {
        int gr = m0 + (wave << 4) + r + (hi << 3);
        if (gr < nrows) {
          float val = cfr[nt][r];
          if (BIAS_RELU) { val += bv; val = val > 0.f ? val : 0.f; }
          if (OUT_BF16) ((unsigned short*)Yv)[(size_t)gr * 64 + n] = f2bf(val);
          else          ((float*)Yv)[(size_t)gr * 64 + n] = val;
        }
      }
    }
  }
}

// el/er: out[n,h] = sum_f P[n, h*F+f] * a[h,f]
__global__ void attn_dot(const float* __restrict__ P, const float* __restrict__ av,
                         float* __restrict__ out, int N, int H, int F) {
  int t = blockIdx.x * blockDim.x + threadIdx.x;
  if (t >= N * H) return;
  int n = t / H, h = t - n * H;
  const float* p = P + (size_t)n * 64 + h * F;
  const float* a = av + h * F;
  float s = 0.f;
  for (int f = 0; f < F; ++f) s = fmaf(p[f], a[f], s);
  out[t] = s;
}

__global__ void fill_i32(int* p, int v, int n) {
  int t = blockIdx.x * blockDim.x + threadIdx.x;
  if (t < n) p[t] = v;
}
__global__ void fill_f32(float* p, float v, int n) {
  int t = blockIdx.x * blockDim.x + threadIdx.x;
  if (t < n) p[t] = v;
}
// acc[r*64+c] = b1[c] (+ b2[c])   (per-etype GAT bias, pre-summed)
__global__ void init_bias2(float* __restrict__ acc, const float* __restrict__ b1,
                           const float* __restrict__ b2, int rows) {
  int t = blockIdx.x * blockDim.x + threadIdx.x;
  if (t >= rows * 64) return;
  int c = t & 63;
  float v = b1[c];
  if (b2) v += b2[c];
  acc[t] = v;
}

template<int H>
__global__ void edge_max_k(const int* __restrict__ src, const int* __restrict__ dst,
                           const float* __restrict__ el, const float* __restrict__ er,
                           int* __restrict__ m, int E) {
  int t = blockIdx.x * blockDim.x + threadIdx.x;
  if (t >= E) return;
  int s = src[t], d = dst[t];
#pragma unroll
  for (int h = 0; h < H; ++h) {
    float e = el[s * H + h] + er[d * H + h];
    e = e > 0.f ? e : 0.2f * e;                 // leaky_relu(0.2)
    atomicMax(&m[d * H + h], f2oi(e));
  }
}

template<int H>
__global__ void edge_sum_k(const int* __restrict__ src, const int* __restrict__ dst,
                           const float* __restrict__ el, const float* __restrict__ er,
                           const int* __restrict__ m, float* __restrict__ ssum, int E) {
  int t = blockIdx.x * blockDim.x + threadIdx.x;
  if (t >= E) return;
  int s = src[t], d = dst[t];
#pragma unroll
  for (int h = 0; h < H; ++h) {
    float e = el[s * H + h] + er[d * H + h];
    e = e > 0.f ? e : 0.2f * e;
    float ex = __expf(e - oi2f(m[d * H + h]));
    atomicAdd(&ssum[d * H + h], ex);
  }
}

// 4 lanes per edge; lane j handles features [j*16, j*16+16)  (head j when H==4)
template<int H>
__global__ void edge_scat_k(const int* __restrict__ src, const int* __restrict__ dst,
                            const float* __restrict__ el, const float* __restrict__ er,
                            const int* __restrict__ m, const float* __restrict__ ssum,
                            const float* __restrict__ P, float* __restrict__ acc, int E) {
  int t = blockIdx.x * blockDim.x + threadIdx.x;
  int e = t >> 2, j = t & 3;
  if (e >= E) return;
  int s = src[e], d = dst[e];
  int h = (H == 4) ? j : 0;
  float x = el[s * H + h] + er[d * H + h];
  x = x > 0.f ? x : 0.2f * x;
  float a = __expf(x - oi2f(m[d * H + h])) / ssum[d * H + h];
  const float4* ps = (const float4*)(P + (size_t)s * 64 + j * 16);
  float* pd = acc + (size_t)d * 64 + j * 16;
#pragma unroll
  for (int q = 0; q < 4; ++q) {
    float4 v = ps[q];
    atomicAdd(pd + 4 * q + 0, v.x * a);
    atomicAdd(pd + 4 * q + 1, v.y * a);
    atomicAdd(pd + 4 * q + 2, v.z * a);
    atomicAdd(pd + 4 * q + 3, v.w * a);
  }
}

__global__ void cvt_bf16_k(const float* __restrict__ x, unsigned short* __restrict__ y, int n) {
  int t = blockIdx.x * blockDim.x + threadIdx.x;
  if (t < n) y[t] = f2bf(x[t]);
}

// one wave32 per 64-float row: squared-sum shuffle reduction then scale
__global__ void l2norm_k(const float* __restrict__ x, float* __restrict__ y, int rows) {
  int gt = blockIdx.x * blockDim.x + threadIdx.x;
  int row = gt >> 5, lane = gt & 31;
  if (row >= rows) return;
  const float2* xr = (const float2*)(x + (size_t)row * 64);
  float2 v = xr[lane];
  float ss = v.x * v.x + v.y * v.y;
#pragma unroll
  for (int o = 16; o > 0; o >>= 1) ss += __shfl_xor(ss, o, 32);
  float nrm = sqrtf(ss);
  float inv = 1.f / (nrm > 1e-12f ? nrm : 1e-12f);
  float2 o2; o2.x = v.x * inv; o2.y = v.y * inv;
  ((float2*)(y + (size_t)row * 64))[lane] = o2;
}

// ---------------------------------------------------------------------------
// Host orchestration
// ---------------------------------------------------------------------------
struct EtW {
  const unsigned short* Wb;   // pre-converted bf16 weight [64,64]
  const float *al, *ar, *b;
};

static inline int cdiv(long long a, int b) { return (int)((a + b - 1) / b); }

static void run_gat(hipStream_t st, int H, int F,
                    const EtW& wri, const EtW& cit, const EtW& wb,
                    const unsigned short* hbA, const unsigned short* hbP,
                    float* PA, float* PP, float* PD,
                    float* el, float* er, int* mb, float* sb,
                    float* accP, float* accA,
                    const int* w_src, const int* w_dst,
                    const int* wb_src, const int* wb_dst,
                    const int* c_src, const int* c_dst)
{
  const int B = 256;
  const int OI_NINF = (int)0x807FFFFF;           // f2oi(-inf)

  init_bias2<<<cdiv((long long)NP * 64, B), B, 0, st>>>(accP, wri.b, cit.b, NP);
  init_bias2<<<cdiv((long long)NA * 64, B), B, 0, st>>>(accA, wb.b, nullptr, NA);

  auto gemm64 = [&](const unsigned short* X, const unsigned short* Wb2, float* Y, int n) {
    gemm_wmma<64, true, false, false><<<cdiv(n, 64), 128, 0, st>>>(X, Wb2, nullptr, Y, n);
  };
  auto attn = [&](const float* P, const float* a, float* o, int n) {
    attn_dot<<<cdiv((long long)n * H, B), B, 0, st>>>(P, a, o, n, H, F);
  };
  auto edges = [&](const int* s, const int* d, const float* P, float* acc, int E, int nd) {
    fill_i32<<<cdiv((long long)nd * H, B), B, 0, st>>>(mb, OI_NINF, nd * H);
    fill_f32<<<cdiv((long long)nd * H, B), B, 0, st>>>(sb, 0.f, nd * H);
    if (H == 4) {
      edge_max_k<4><<<cdiv(E, B), B, 0, st>>>(s, d, el, er, mb, E);
      edge_sum_k<4><<<cdiv(E, B), B, 0, st>>>(s, d, el, er, mb, sb, E);
      edge_scat_k<4><<<cdiv((long long)E * 4, B), B, 0, st>>>(s, d, el, er, mb, sb, P, acc, E);
    } else {
      edge_max_k<1><<<cdiv(E, B), B, 0, st>>>(s, d, el, er, mb, E);
      edge_sum_k<1><<<cdiv(E, B), B, 0, st>>>(s, d, el, er, mb, sb, E);
      edge_scat_k<1><<<cdiv((long long)E * 4, B), B, 0, st>>>(s, d, el, er, mb, sb, P, acc, E);
    }
  };

  // writes: author -> paper
  gemm64(hbA, wri.Wb, PA, NA);
  gemm64(hbP, wri.Wb, PD, NP);
  attn(PA, wri.al, el, NA);
  attn(PD, wri.ar, er, NP);
  edges(w_src, w_dst, PA, accP, EW, NP);

  // cites: paper -> paper (src proj == dst proj, single GEMM)
  gemm64(hbP, cit.Wb, PP, NP);
  attn(PP, cit.al, el, NP);
  attn(PP, cit.ar, er, NP);
  edges(c_src, c_dst, PP, accP, EC, NP);

  // written_by: paper -> author
  gemm64(hbP, wb.Wb, PP, NP);
  gemm64(hbA, wb.Wb, PA, NA);
  attn(PP, wb.al, el, NP);
  attn(PA, wb.ar, er, NA);
  edges(wb_src, wb_dst, PP, accA, EW, NA);
}

extern "C" void kernel_launch(void* const* d_in, const int* in_sizes, int n_in,
                              void* d_out, int out_size, void* d_ws, size_t ws_size,
                              hipStream_t stream)
{
  (void)in_sizes; (void)n_in; (void)out_size; (void)ws_size;
  const float* feat_a = (const float*)d_in[0];
  const float* feat_p = (const float*)d_in[1];
  const float* fmWa   = (const float*)d_in[2];
  const float* fmba   = (const float*)d_in[3];
  const float* fmWp   = (const float*)d_in[4];
  const float* fmbp   = (const float*)d_in[5];
  const int* w_src  = (const int*)d_in[30];
  const int* w_dst  = (const int*)d_in[31];
  const int* wb_src = (const int*)d_in[32];
  const int* wb_dst = (const int*)d_in[33];
  const int* c_src  = (const int*)d_in[34];
  const int* c_dst  = (const int*)d_in[35];

  // workspace carve (256B aligned); every buffer fully rewritten each call
  char* w = (char*)d_ws; size_t off = 0;
  auto carve = [&](size_t bytes) -> void* {
    void* p = w + off; off = (off + bytes + 255) & ~(size_t)255; return p;
  };
  unsigned short* hbA = (unsigned short*)carve((size_t)NA * 64 * 2);
  unsigned short* hbP = (unsigned short*)carve((size_t)NP * 64 * 2);
  float* PA   = (float*)carve((size_t)NA * 64 * 4);
  float* PP   = (float*)carve((size_t)NP * 64 * 4);
  float* PD   = (float*)carve((size_t)NP * 64 * 4);
  float* el   = (float*)carve((size_t)NP * 4 * 4);
  float* er   = (float*)carve((size_t)NP * 4 * 4);
  int*   mb   = (int*)  carve((size_t)NP * 4 * 4);
  float* sb   = (float*)carve((size_t)NP * 4 * 4);
  float* accP = (float*)carve((size_t)NP * 64 * 4);
  float* accA = (float*)carve((size_t)NA * 64 * 4);
  // bf16 weight buffers
  unsigned short* fmWa_b = (unsigned short*)carve((size_t)64 * 128 * 2);
  unsigned short* fmWp_b = (unsigned short*)carve((size_t)64 * 256 * 2);
  unsigned short* gWb[6];
  for (int i = 0; i < 6; ++i) gWb[i] = (unsigned short*)carve((size_t)64 * 64 * 2);

  // one-time per-launch weight conversion f32 -> bf16
  cvt_bf16_k<<<cdiv(64 * 128, 256), 256, 0, stream>>>(fmWa, fmWa_b, 64 * 128);
  cvt_bf16_k<<<cdiv(64 * 256, 256), 256, 0, stream>>>(fmWp, fmWp_b, 64 * 256);
  const int wIdx[6] = {6, 10, 14, 18, 22, 26};   // g1w,g2w,g1wb,g2wb,g1c,g2c
  for (int i = 0; i < 6; ++i)
    cvt_bf16_k<<<cdiv(64 * 64, 256), 256, 0, stream>>>(
        (const float*)d_in[wIdx[i]], gWb[i], 64 * 64);

  auto et = [&](int base, const unsigned short* Wb2) -> EtW {
    return EtW{Wb2, (const float*)d_in[base + 1],
               (const float*)d_in[base + 2], (const float*)d_in[base + 3]};
  };
  EtW g1w  = et(6,  gWb[0]), g2w  = et(10, gWb[1]);
  EtW g1wb = et(14, gWb[2]), g2wb = et(18, gWb[3]);
  EtW g1c  = et(22, gWb[4]), g2c  = et(26, gWb[5]);

  // feature mapping: relu(X @ W^T + b) -> bf16 hidden
  gemm_wmma<128, false, true, true><<<cdiv(NA, 64), 128, 0, stream>>>(feat_a, fmWa_b, fmba, hbA, NA);
  gemm_wmma<256, false, true, true><<<cdiv(NP, 64), 128, 0, stream>>>(feat_p, fmWp_b, fmbp, hbP, NP);

  // layer 1: 4 heads x 16
  run_gat(stream, 4, 16, g1w, g1c, g1wb, hbA, hbP, PA, PP, PD, el, er, mb, sb,
          accP, accA, w_src, w_dst, wb_src, wb_dst, c_src, c_dst);
  cvt_bf16_k<<<cdiv((long long)NA * 64, 256), 256, 0, stream>>>(accA, hbA, NA * 64);
  cvt_bf16_k<<<cdiv((long long)NP * 64, 256), 256, 0, stream>>>(accP, hbP, NP * 64);

  // layer 2: 1 head x 64
  run_gat(stream, 1, 64, g2w, g2c, g2wb, hbA, hbP, PA, PP, PD, el, er, mb, sb,
          accP, accA, w_src, w_dst, wb_src, wb_dst, c_src, c_dst);

  // L2 normalize -> output (author block then paper block)
  float* outA = (float*)d_out;
  float* outP = outA + (size_t)NA * 64;
  l2norm_k<<<cdiv((long long)NA * 32, 256), 256, 0, stream>>>(accA, outA, NA);
  l2norm_k<<<cdiv((long long)NP * 32, 256), 256, 0, stream>>>(accP, outP, NP);
}
